// GraphProcess_17600775979613
// MI455X (gfx1250) — matrix operand order repaired and checked by reading, not compiled, not run on hardware
//
#include <hip/hip_runtime.h>
#include <hip/hip_bf16.h>
#include <math.h>

// ---------------- model constants ----------------
#define Bb    2
#define Cc    128
#define Hh    128
#define Ww    128
#define NBLKc 64
#define BNODc 16
#define Nn    1024
#define NPc   256
#define ROWS  (Bb * Nn)   // 2048

typedef __attribute__((ext_vector_type(16))) _Float16 v16h;
typedef __attribute__((ext_vector_type(8)))  float    v8f;

// ---------------- WMMA fragment helpers (gfx1250 wave32 layouts) ----------------
// A (16x32 f16, row-major source, row stride ld halfs):
//   lane l: row = l&15, Kbase = (l>>4)*8 ; halfs[0..7]=K[Kbase..+7], halfs[8..15]=K[Kbase+16..+23]
__device__ __forceinline__ v16h load_a16(const _Float16* tile, int ld) {
  int l = threadIdx.x & 31;
  const _Float16* r = tile + (size_t)(l & 15) * ld + ((l >> 4) * 8);
  v16h a;
#pragma unroll
  for (int i = 0; i < 8; ++i) { a[i] = r[i]; a[i + 8] = r[16 + i]; }
  return a;
}

// B (32x16 f16) given B^T row-major (BT[n][k], row stride ld halfs):
//   lane l: col = l&15, Kbase = (l>>4)*8 ; halfs[0..7]=BT[col][Kbase..+7], halfs[8..15]=BT[col][Kbase+16..+23]
__device__ __forceinline__ v16h load_bt16(const _Float16* tileT, int ld) {
  int l = threadIdx.x & 31;
  const _Float16* r = tileT + (size_t)(l & 15) * ld + ((l >> 4) * 8);
  v16h b;
#pragma unroll
  for (int i = 0; i < 8; ++i) { b[i] = r[i]; b[i + 8] = r[16 + i]; }
  return b;
}

__device__ __forceinline__ v8f wmma16(v16h a, v16h b, v8f c) {
  return __builtin_amdgcn_wmma_f32_16x16x32_f16(false, a, false, b, (short)0, c, false, false);
}
// D fragment mapping: reg r, lane l  ->  (M = r + 8*(l>>4), N = l&15)

__device__ __forceinline__ float sigf(float x) { return 1.0f / (1.0f + __expf(-x)); }

// ---------------- K: init (zero global-sumsq accumulators) ----------------
__global__ void k_init(float* gss) {
  if (threadIdx.x < Bb) gss[threadIdx.x] = 0.0f;
}

// ---------------- K: transpose + f16-convert weight matrices ----------------
__global__ void k_weights(const float* __restrict__ Wn, const float* __restrict__ Wr,
                          const float* __restrict__ Wa, const float* __restrict__ Wf,
                          _Float16* TN, _Float16* TR, _Float16* TA, _Float16* TF) {
  int idx = blockIdx.x * 256 + threadIdx.x;
  int which = blockIdx.y;
  if (which < 3) {
    if (idx < 3 * Cc * Cc) {
      int s = idx / (Cc * Cc), r = idx % (Cc * Cc);
      int ci = r / Cc, co = r % Cc;
      const float* src = (which == 0) ? Wn : (which == 1) ? Wr : Wa;
      _Float16* dst = (which == 0) ? TN : (which == 1) ? TR : TA;
      dst[((size_t)s * Cc + co) * Cc + ci] = (_Float16)src[idx];
    }
  } else {
    if (idx < Cc * Cc) {
      int ci = idx / Cc, co = idx % Cc;
      TF[(size_t)co * Cc + ci] = (_Float16)Wf[idx];
    }
  }
}

// ---------------- K: graph projection (per batch,block) ----------------
__global__ __launch_bounds__(256) void k_project(const float* __restrict__ x,
                                                 const float* __restrict__ anchor,
                                                 const float* __restrict__ sigma_raw,
                                                 float* __restrict__ Q,
                                                 float* __restrict__ nodes) {
  __shared__ float anc_s[BNODc * Cc];
  __shared__ float isg_s[BNODc * Cc];
  __shared__ float sa_s[BNODc * NPc];
  __shared__ float Ssh[BNODc];
  int blk = blockIdx.x, b = blockIdx.y, t = threadIdx.x;
  int by = blk >> 3, bx = blk & 7;

  for (int i = t; i < BNODc * Cc; i += 256) {
    int nn = i >> 7, c = i & 127;
    int ng = blk * BNODc + nn;
    anc_s[i] = anchor[(size_t)ng * Cc + c];
    isg_s[i] = 1.0f + __expf(-sigma_raw[(size_t)ng * Cc + c]);  // 1/sigmoid(z)
  }
  __syncthreads();

  int hb = t >> 4, wb = t & 15;
  int h = by * 16 + hb, w = bx * 16 + wb;
  const float* xp = x + (size_t)b * Cc * Hh * Ww + (size_t)h * Ww + w;

  float acc[BNODc];
#pragma unroll
  for (int nn = 0; nn < BNODc; ++nn) acc[nn] = 0.0f;
  for (int c = 0; c < Cc; ++c) {
    float xc = xp[(size_t)c * Hh * Ww];
#pragma unroll
    for (int nn = 0; nn < BNODc; ++nn) {
      float r = (xc - anc_s[nn * Cc + c]) * isg_s[nn * Cc + c];
      acc[nn] += r * r;
    }
  }
  float mx = -1e30f;
#pragma unroll
  for (int nn = 0; nn < BNODc; ++nn) { acc[nn] = -0.5f * acc[nn]; mx = fmaxf(mx, acc[nn]); }
  float se = 0.0f;
#pragma unroll
  for (int nn = 0; nn < BNODc; ++nn) { acc[nn] = __expf(acc[nn] - mx); se += acc[nn]; }
  float inv = 1.0f / se;
#pragma unroll
  for (int nn = 0; nn < BNODc; ++nn) {
    float sa = acc[nn] * inv;
    sa_s[nn * NPc + t] = sa;
    Q[((size_t)(b * Nn + blk * BNODc + nn)) * NPc + t] = sa;
  }
  __syncthreads();
  if (t < BNODc) {
    float s = 0.0f;
    for (int p = 0; p < NPc; ++p) s += sa_s[t * NPc + p];
    Ssh[t] = s;
  }
  __syncthreads();

  // nodes[n,c] = (sum_p sa*pix - anc*S) * isig / (S + eps)
  int nn = t >> 4, cg = t & 15;
  float S = Ssh[nn];
#pragma unroll
  for (int cc = 0; cc < 8; ++cc) {
    int c = cg * 8 + cc;
    const float* xpc = x + (((size_t)b * Cc + c) * Hh + by * 16) * Ww + bx * 16;
    float s = 0.0f;
    for (int p = 0; p < NPc; ++p)
      s += sa_s[nn * NPc + p] * xpc[(p >> 4) * Ww + (p & 15)];
    float nv = (s - anc_s[nn * Cc + c] * S) * isg_s[nn * Cc + c] / (S + 1e-9f);
    nodes[((size_t)(b * Nn + blk * BNODc + nn)) * Cc + c] = nv;
  }
}

// ---------------- K: per-node L2 normalize + accumulate global sumsq ----------------
__global__ __launch_bounds__(128) void k_norm1(float* __restrict__ nodes, float* gss) {
  __shared__ float red[128];
  int n = blockIdx.x, b = blockIdx.y, t = threadIdx.x;
  size_t base = ((size_t)(b * Nn + n)) * Cc;
  float v = nodes[base + t];
  red[t] = v * v;
  __syncthreads();
  for (int st = 64; st > 0; st >>= 1) { if (t < st) red[t] += red[t + st]; __syncthreads(); }
  float ss = red[0];
  float inv = 1.0f / fmaxf(sqrtf(ss), 1e-12f);
  nodes[base + t] = v * inv;
  if (t == 0) atomicAdd(&gss[b], ss * inv * inv);
}

// ---------------- K: global scale + channel renorm -> feats (f32 + f16) ----------------
__global__ __launch_bounds__(128) void k_norm2(const float* __restrict__ nodes, const float* gss,
                                               float* __restrict__ feats, _Float16* __restrict__ f16) {
  __shared__ float red[128];
  int n = blockIdx.x, b = blockIdx.y, t = threadIdx.x;
  size_t base = ((size_t)(b * Nn + n)) * Cc;
  float gin = 1.0f / fmaxf(sqrtf(gss[b]), 1e-12f);
  float v2 = nodes[base + t] * gin;
  red[t] = v2 * v2;
  __syncthreads();
  for (int st = 64; st > 0; st >>= 1) { if (t < st) red[t] += red[t + st]; __syncthreads(); }
  float inv = 1.0f / fmaxf(sqrtf(red[0]), 1e-12f);
  float fv = v2 * inv;
  feats[base + t] = fv;
  f16[base + t] = (_Float16)fv;
}

// ---------------- K: gate projections gi/gj ----------------
__global__ __launch_bounds__(256) void k_gate(const float* __restrict__ feats,
                                              const float* __restrict__ wsrc,
                                              const float* __restrict__ wdst,
                                              float* __restrict__ gi, float* __restrict__ gj) {
  __shared__ float ws_s[Cc], wd_s[Cc];
  int s = blockIdx.x, b = blockIdx.y, t = threadIdx.x;
  if (t < Cc) { ws_s[t] = wsrc[s * Cc + t]; wd_s[t] = wdst[s * Cc + t]; }
  __syncthreads();
  for (int n = t; n < Nn; n += 256) {
    const float* f = feats + ((size_t)(b * Nn + n)) * Cc;
    float a = 0.0f, c2 = 0.0f;
    for (int c = 0; c < Cc; ++c) { float fv = f[c]; a += fv * ws_s[c]; c2 += fv * wd_s[c]; }
    gi[((size_t)(s * Bb + b)) * Nn + n] = a;
    gj[((size_t)(s * Bb + b)) * Nn + n] = c2;
  }
}

// ---------------- K: HnT = (feats @ W_nbr[s])^T, f16  (WMMA) ----------------
__global__ __launch_bounds__(32) void k_hnbr(const _Float16* __restrict__ f16,
                                             const _Float16* __restrict__ TN,
                                             _Float16* __restrict__ HnT) {
  int rt = blockIdx.x, ct = blockIdx.y, s = blockIdx.z;
  int l = threadIdx.x;
  const _Float16* A0 = f16 + (size_t)rt * 16 * Cc;
  const _Float16* BT = TN + ((size_t)s * Cc + ct * 16) * Cc;
  v8f d = {};
#pragma unroll
  for (int kc = 0; kc < 4; ++kc)
    d = wmma16(load_a16(A0 + kc * 32, Cc), load_bt16(BT + kc * 32, Cc), d);
  int ncol = l & 15, mh = (l >> 4) * 8;
  int c = ct * 16 + ncol;
#pragma unroll
  for (int r = 0; r < 8; ++r) {
    int rg = rt * 16 + mh + r;
    int b = rg >> 10, n = rg & 1023;
    HnT[(((size_t)s * Bb + b) * Cc + c) * Nn + n] = (_Float16)d[r];
  }
}

// ---------------- K: fused adjacency + mask + gate + aggregation (WMMA) ----------------
// out = feats@W_root + (mask(adj)*sigmoid(gi+gj+b)) @ (feats@W_nbr)
__global__ __launch_bounds__(32) void k_agg(const _Float16* __restrict__ f16,
                                            const _Float16* __restrict__ TR,
                                            const _Float16* __restrict__ HnT,
                                            const float* __restrict__ gi,
                                            const float* __restrict__ gj,
                                            const float* __restrict__ bgate,
                                            float* __restrict__ Obuf) {
  __shared__ _Float16 wt[16 * 32];
  __shared__ float gi_s[16];
  int it = blockIdx.x, s = blockIdx.y, b = blockIdx.z;
  int l = threadIdx.x;
  int ibase = it * 16;

  const _Float16* Af = f16 + ((size_t)b * Nn + ibase) * Cc;
  v16h afr[4];
#pragma unroll
  for (int kc = 0; kc < 4; ++kc) afr[kc] = load_a16(Af + kc * 32, Cc);
  if (l < 16) gi_s[l] = gi[((size_t)(s * Bb + b)) * Nn + ibase + l];
  __syncthreads();

  float bg = bgate[s];
  float thr = (s == 0) ? 0.7f : ((s == 1) ? 0.5f : 0.3f);

  v8f acc[8];
#pragma unroll
  for (int ct = 0; ct < 8; ++ct) {          // init: feats @ W_root[s]
    v8f d = {};
#pragma unroll
    for (int kc = 0; kc < 4; ++kc)
      d = wmma16(afr[kc], load_bt16(TR + ((size_t)s * Cc + ct * 16) * Cc + kc * 32, Cc), d);
    acc[ct] = d;
  }

  int ncol = l & 15, mh = (l >> 4) * 8;
  const float* gjp = gj + ((size_t)(s * Bb + b)) * Nn;
  const _Float16* HT0 = HnT + (((size_t)s * Bb + b) * Cc) * Nn;

  for (int j = 0; j < Nn; j += 32) {
    const _Float16* Bf = f16 + ((size_t)b * Nn + j) * Cc;
    if (j + 32 < Nn) __builtin_prefetch(Bf + 32 * Cc, 0, 1);  // -> global_prefetch
    v8f d0 = {}, d1 = {};
#pragma unroll
    for (int kc = 0; kc < 4; ++kc) {
      d0 = wmma16(afr[kc], load_bt16(Bf + kc * 32, Cc), d0);
      d1 = wmma16(afr[kc], load_bt16(Bf + 16 * Cc + kc * 32, Cc), d1);
    }
    float gj0 = gjp[j + ncol], gj1 = gjp[j + 16 + ncol];
#pragma unroll
    for (int r = 0; r < 8; ++r) {
      int M = mh + r;
      int rn = ibase + M;
      float gv = gi_s[M];
      {
        int cn = j + ncol;
        float adj = d0[r];
        bool same = (rn >> 4) == (cn >> 4);
        bool m = (s == 0) ? (adj > thr && same) : (s == 1) ? (adj > thr && !same) : (adj > thr);
        float wv = m ? adj : 0.0f;
        wt[M * 32 + ncol] = (_Float16)(wv * sigf(gv + gj0 + bg));
      }
      {
        int cn = j + 16 + ncol;
        float adj = d1[r];
        bool same = (rn >> 4) == (cn >> 4);
        bool m = (s == 0) ? (adj > thr && same) : (s == 1) ? (adj > thr && !same) : (adj > thr);
        float wv = m ? adj : 0.0f;
        wt[M * 32 + 16 + ncol] = (_Float16)(wv * sigf(gv + gj1 + bg));
      }
    }
    __syncthreads();
    v16h wa = load_a16(wt, 32);
    __syncthreads();
    const _Float16* HT = HT0 + j;
#pragma unroll
    for (int ct = 0; ct < 8; ++ct)
      acc[ct] = wmma16(wa, load_bt16(HT + (size_t)ct * 16 * Nn, Nn), acc[ct]);
  }

#pragma unroll
  for (int ct = 0; ct < 8; ++ct)
#pragma unroll
    for (int r = 0; r < 8; ++r) {
      int M = mh + r;
      Obuf[((size_t)s * ROWS + b * Nn + ibase + M) * Cc + ct * 16 + ncol] = acc[ct][r];
    }
}

// ---------------- K: BatchNorm statistics over B*N rows ----------------
__global__ __launch_bounds__(256) void k_bnstats(const float* __restrict__ Obuf,
                                                 float* __restrict__ bnmu, float* __restrict__ bninv) {
  __shared__ float s1[256], s2[256];
  int c = blockIdx.x, s = blockIdx.y, t = threadIdx.x;
  float a = 0.0f, q = 0.0f;
  for (int r = t; r < ROWS; r += 256) {
    float v = Obuf[((size_t)s * ROWS + r) * Cc + c];
    a += v; q += v * v;
  }
  s1[t] = a; s2[t] = q;
  __syncthreads();
  for (int st = 128; st > 0; st >>= 1) {
    if (t < st) { s1[t] += s1[t + st]; s2[t] += s2[t + st]; }
    __syncthreads();
  }
  if (t == 0) {
    float mu = s1[0] / (float)ROWS;
    float var = s2[0] / (float)ROWS - mu * mu;
    bnmu[s * Cc + c] = mu;
    bninv[s * Cc + c] = rsqrtf(var + 1e-5f);
  }
}

// ---------------- K: apply BN + residual -> h (f32 + f16) ----------------
__global__ __launch_bounds__(256) void k_bnapply(const float* __restrict__ Obuf,
                                                 const float* __restrict__ bnmu,
                                                 const float* __restrict__ bninv,
                                                 const float* __restrict__ gamma,
                                                 const float* __restrict__ beta,
                                                 const float* __restrict__ feats,
                                                 float* __restrict__ hbuf, _Float16* __restrict__ h16) {
  size_t idx = (size_t)blockIdx.x * 256 + threadIdx.x;
  if (idx >= (size_t)3 * ROWS * Cc) return;
  int s = (int)(idx / ((size_t)ROWS * Cc));
  size_t rem = idx % ((size_t)ROWS * Cc);
  int r = (int)(rem / Cc), c = (int)(rem % Cc);
  float o = Obuf[idx];
  float hv = gamma[s * Cc + c] * (o - bnmu[s * Cc + c]) * bninv[s * Cc + c] + beta[s * Cc + c]
           + feats[(size_t)r * Cc + c];
  hbuf[idx] = hv;
  h16[idx] = (_Float16)hv;
}

// ---------------- K: node attention  y = relu(sigmoid(h@attW+b) * h)  (WMMA) ----------------
__global__ __launch_bounds__(32) void k_att(const _Float16* __restrict__ h16,
                                            const float* __restrict__ hbuf,
                                            const _Float16* __restrict__ TA,
                                            const float* __restrict__ att_b,
                                            float* __restrict__ ybuf) {
  int rt = blockIdx.x, ct = blockIdx.y, s = blockIdx.z;
  int l = threadIdx.x;
  const _Float16* A0 = h16 + ((size_t)s * ROWS + rt * 16) * Cc;
  const _Float16* BT = TA + ((size_t)s * Cc + ct * 16) * Cc;
  v8f d = {};
#pragma unroll
  for (int kc = 0; kc < 4; ++kc)
    d = wmma16(load_a16(A0 + kc * 32, Cc), load_bt16(BT + kc * 32, Cc), d);
  int ncol = l & 15, mh = (l >> 4) * 8;
  int c = ct * 16 + ncol;
  float bv = att_b[s * Cc + c];
#pragma unroll
  for (int r = 0; r < 8; ++r) {
    int rg = rt * 16 + mh + r;
    size_t idx = ((size_t)s * ROWS + rg) * Cc + c;
    float yv = sigf(d[r] + bv) * hbuf[idx];
    ybuf[idx] = (yv > 0.0f) ? yv : 0.0f;
  }
}

// ---------------- K: global max pool over nodes ----------------
__global__ __launch_bounds__(128) void k_pool(const float* __restrict__ ybuf, float* __restrict__ pool) {
  int s = blockIdx.x, b = blockIdx.y, c = threadIdx.x;
  float m = -1e30f;
  for (int n = 0; n < Nn; ++n)
    m = fmaxf(m, ybuf[((size_t)s * ROWS + b * Nn + n) * Cc + c]);
  pool[(size_t)(s * Bb + b) * Cc + c] = m;
}

// ---------------- K: small MLP + scale logits ----------------
__global__ __launch_bounds__(128) void k_mlp(const float* __restrict__ pool,
                                             const float* __restrict__ W1, const float* __restrict__ b1,
                                             const float* __restrict__ W2, const float* __restrict__ b2,
                                             const float* __restrict__ lineA_w, const float* __restrict__ lineA_b,
                                             float* __restrict__ mlpp, float* __restrict__ avec) {
  __shared__ float p_s[Cc], r1_s[Cc], red[Cc];
  int sb = blockIdx.x, t = threadIdx.x;
  int s = sb >> 1, b = sb & 1;
  p_s[t] = pool[(size_t)sb * Cc + t];
  __syncthreads();
  float a1 = b1[s * Cc + t];
  for (int k = 0; k < Cc; ++k) a1 += p_s[k] * W1[((size_t)s * Cc + k) * Cc + t];
  r1_s[t] = (a1 > 0.0f) ? a1 : 0.0f;
  __syncthreads();
  float a2 = b2[s * Cc + t];
  for (int k = 0; k < Cc; ++k) a2 += r1_s[k] * W2[((size_t)s * Cc + k) * Cc + t];
  float pl = sigf(a2);
  mlpp[(size_t)sb * Cc + t] = pl;
  red[t] = pl * lineA_w[s * Cc + t];
  __syncthreads();
  for (int st = 64; st > 0; st >>= 1) { if (t < st) red[t] += red[t + st]; __syncthreads(); }
  if (t == 0) avec[b * 3 + s] = red[0] + lineA_b[s];
}

// ---------------- K: scale softmax + fuse -> fused (f16) ----------------
__global__ __launch_bounds__(128) void k_fused(const float* __restrict__ ybuf,
                                               const float* __restrict__ mlpp,
                                               const float* __restrict__ avec,
                                               _Float16* __restrict__ fused16) {
  int bn = blockIdx.x, c = threadIdx.x;
  int b = bn >> 10;
  float a0 = avec[b * 3 + 0], a1 = avec[b * 3 + 1], a2 = avec[b * 3 + 2];
  float m = fmaxf(a0, fmaxf(a1, a2));
  float e0 = __expf(a0 - m), e1 = __expf(a1 - m), e2 = __expf(a2 - m);
  float inv = 1.0f / (e0 + e1 + e2);
  float A[3] = { e0 * inv, e1 * inv, e2 * inv };
  float fv = 0.0f;
#pragma unroll
  for (int s = 0; s < 3; ++s)
    fv += A[s] * mlpp[(size_t)(s * Bb + b) * Cc + c]
               * ybuf[((size_t)s * ROWS + bn) * Cc + c];
  fused16[(size_t)bn * Cc + c] = (_Float16)fv;
}

// ---------------- K: yfin = fused @ lineFu_W + b  (WMMA) ----------------
__global__ __launch_bounds__(32) void k_linefu(const _Float16* __restrict__ fused16,
                                               const _Float16* __restrict__ TF,
                                               const float* __restrict__ fu_b,
                                               float* __restrict__ yfin) {
  int rt = blockIdx.x, ct = blockIdx.y;
  int l = threadIdx.x;
  const _Float16* A0 = fused16 + (size_t)rt * 16 * Cc;
  const _Float16* BT = TF + (size_t)ct * 16 * Cc;
  v8f d = {};
#pragma unroll
  for (int kc = 0; kc < 4; ++kc)
    d = wmma16(load_a16(A0 + kc * 32, Cc), load_bt16(BT + kc * 32, Cc), d);
  int ncol = l & 15, mh = (l >> 4) * 8;
  int c = ct * 16 + ncol;
  float bv = fu_b[c];
#pragma unroll
  for (int r = 0; r < 8; ++r) {
    int rg = rt * 16 + mh + r;
    yfin[(size_t)rg * Cc + c] = d[r] + bv;
  }
}

// ---------------- K: reprojection + residual ----------------
__global__ __launch_bounds__(256) void k_reproj(const float* __restrict__ yfin,
                                                const float* __restrict__ Q,
                                                const float* __restrict__ x,
                                                float* __restrict__ out) {
  __shared__ float Zs[BNODc][Cc];
  __shared__ float Qs[BNODc][NPc];
  int blk = blockIdx.x, b = blockIdx.y, t = threadIdx.x;
  for (int i = t; i < BNODc * Cc; i += 256) {
    int nn = i >> 7, c = i & 127;
    Zs[nn][c] = yfin[((size_t)b * Nn + blk * BNODc + nn) * Cc + c];
  }
  for (int i = t; i < BNODc * NPc; i += 256) {
    int nn = i >> 8, p = i & 255;
    Qs[nn][p] = Q[((size_t)b * Nn + blk * BNODc + nn) * NPc + p];
  }
  __syncthreads();
  int by = blk >> 3, bx = blk & 7;
  int h = by * 16 + (t >> 4), w = bx * 16 + (t & 15);
  for (int c = 0; c < Cc; ++c) {
    float s = 0.0f;
#pragma unroll
    for (int nn = 0; nn < BNODc; ++nn) s += Zs[nn][c] * Qs[nn][t];
    size_t xi = (((size_t)b * Cc + c) * Hh + h) * Ww + w;
    out[xi] = s + x[xi];
  }
}

// ---------------- host launch ----------------
extern "C" void kernel_launch(void* const* d_in, const int* in_sizes, int n_in,
                              void* d_out, int out_size, void* d_ws, size_t ws_size,
                              hipStream_t stream) {
  const float* x         = (const float*)d_in[0];
  const float* anchor    = (const float*)d_in[1];
  const float* sigma_raw = (const float*)d_in[2];
  const float* W_root    = (const float*)d_in[3];
  const float* W_nbr     = (const float*)d_in[4];
  const float* wgs       = (const float*)d_in[5];
  const float* wgd       = (const float*)d_in[6];
  const float* b_gate    = (const float*)d_in[7];
  const float* bn_gamma  = (const float*)d_in[8];
  const float* bn_beta   = (const float*)d_in[9];
  const float* att_W     = (const float*)d_in[10];
  const float* att_b     = (const float*)d_in[11];
  const float* mlp_W1    = (const float*)d_in[12];
  const float* mlp_b1    = (const float*)d_in[13];
  const float* mlp_W2    = (const float*)d_in[14];
  const float* mlp_b2    = (const float*)d_in[15];
  const float* lineA_w   = (const float*)d_in[16];
  const float* lineA_b   = (const float*)d_in[17];
  const float* lineFu_W  = (const float*)d_in[18];
  const float* lineFu_b  = (const float*)d_in[19];
  float* out = (float*)d_out;

  // workspace bump allocator
  char* w = (char*)d_ws;
  size_t off = 0;
  auto alloc = [&](size_t bytes) -> void* {
    void* p = (void*)(w + off);
    off = (off + bytes + 255) & ~(size_t)255;
    return p;
  };
  float*    Qf     = (float*)   alloc((size_t)Bb * Nn * NPc * 4);     // 2 MB
  float*    nodes  = (float*)   alloc((size_t)Bb * Nn * Cc * 4);      // 1 MB (normalized in place)
  float*    gss    = (float*)   alloc(256);
  float*    feats  = (float*)   alloc((size_t)Bb * Nn * Cc * 4);      // 1 MB
  _Float16* f16    = (_Float16*)alloc((size_t)Bb * Nn * Cc * 2);
  _Float16* TN     = (_Float16*)alloc((size_t)3 * Cc * Cc * 2);
  _Float16* TR     = (_Float16*)alloc((size_t)3 * Cc * Cc * 2);
  _Float16* TA     = (_Float16*)alloc((size_t)3 * Cc * Cc * 2);
  _Float16* TF     = (_Float16*)alloc((size_t)Cc * Cc * 2);
  float*    gi     = (float*)   alloc((size_t)3 * Bb * Nn * 4);
  float*    gj     = (float*)   alloc((size_t)3 * Bb * Nn * 4);
  _Float16* HnT    = (_Float16*)alloc((size_t)3 * Bb * Cc * Nn * 2);  // 1.5 MB
  float*    Obuf   = (float*)   alloc((size_t)3 * ROWS * Cc * 4);     // 3 MB
  float*    bnmu   = (float*)   alloc((size_t)3 * Cc * 4);
  float*    bninv  = (float*)   alloc((size_t)3 * Cc * 4);
  float*    hbuf   = (float*)   alloc((size_t)3 * ROWS * Cc * 4);     // 3 MB
  _Float16* h16    = (_Float16*)alloc((size_t)3 * ROWS * Cc * 2);     // 1.5 MB
  float*    pool   = (float*)   alloc((size_t)3 * Bb * Cc * 4);
  float*    mlpp   = (float*)   alloc((size_t)3 * Bb * Cc * 4);
  float*    avec   = (float*)   alloc(256);
  // safe aliases (previous contents dead at use time)
  float*    ybuf    = Obuf;
  _Float16* fused16 = h16;
  float*    yfin    = hbuf;

  k_init   <<<1, 32, 0, stream>>>(gss);
  k_weights<<<dim3(192, 4), 256, 0, stream>>>(W_nbr, W_root, att_W, lineFu_W, TN, TR, TA, TF);
  k_project<<<dim3(NBLKc, Bb), 256, 0, stream>>>(x, anchor, sigma_raw, Qf, nodes);
  k_norm1  <<<dim3(Nn, Bb), 128, 0, stream>>>(nodes, gss);
  k_norm2  <<<dim3(Nn, Bb), 128, 0, stream>>>(nodes, gss, feats, f16);
  k_gate   <<<dim3(3, Bb), 256, 0, stream>>>(feats, wgs, wgd, gi, gj);
  k_hnbr   <<<dim3(ROWS / 16, Cc / 16, 3), 32, 0, stream>>>(f16, TN, HnT);
  k_agg    <<<dim3(Nn / 16, 3, Bb), 32, 0, stream>>>(f16, TR, HnT, gi, gj, b_gate, Obuf);
  k_bnstats<<<dim3(Cc, 3), 256, 0, stream>>>(Obuf, bnmu, bninv);
  k_bnapply<<<(3 * ROWS * Cc + 255) / 256, 256, 0, stream>>>(Obuf, bnmu, bninv, bn_gamma, bn_beta,
                                                             feats, hbuf, h16);
  k_att    <<<dim3(ROWS / 16, Cc / 16, 3), 32, 0, stream>>>(h16, hbuf, TA, att_b, ybuf);
  k_pool   <<<dim3(3, Bb), 128, 0, stream>>>(ybuf, pool);
  k_mlp    <<<6, 128, 0, stream>>>(pool, mlp_W1, mlp_b1, mlp_W2, mlp_b2, lineA_w, lineA_b, mlpp, avec);
  k_fused  <<<ROWS, 128, 0, stream>>>(ybuf, mlpp, avec, fused16);
  k_linefu <<<dim3(ROWS / 16, Cc / 16), 32, 0, stream>>>(fused16, TF, lineFu_b, yfin);
  k_reproj <<<dim3(NBLKc, Bb), 256, 0, stream>>>(yfin, Qf, x, out);

  (void)in_sizes; (void)n_in; (void)out_size; (void)ws_size;
}
